// BilateralLayer_83837761618128
// MI455X (gfx1250) — matrix operand order repaired and checked by reading, not compiled, not run on hardware
//
#include <hip/hip_runtime.h>
#include <hip/hip_bf16.h>

// ---------------------------------------------------------------------------
// BilateralLayer for MI455X (gfx1250, wave32, WMMA)
//  - CNN convs as implicit-GEMM v_wmma_f32_16x16x32_f16 (M=16 out-ch, N=16 px)
//    Compile-time conv configs -> unrolled WMMA chains; LDS staging is
//    branchless (clamped loads + select) and pair-packed (ds_store_b32);
//    fragment gathers are contiguous 16B vectors (ds_load_b128).
//  - Bilateral hashed-lattice solver: L2-resident (each T=2^20 array is 4-12MB,
//    192MB L2 holds the whole CG state), so plain global kernels are optimal.
// ---------------------------------------------------------------------------

typedef __attribute__((ext_vector_type(16))) _Float16 v16h;
typedef __attribute__((ext_vector_type(8)))  _Float16 v8h;
typedef __attribute__((ext_vector_type(2)))  _Float16 v2h;
typedef __attribute__((ext_vector_type(8)))  float    v8f;

#define TSIZE   (1 << 20)
#define TMASK   (TSIZE - 1)
#define LAMBDA  200.0f
#define EPS12   1e-12f

__device__ __forceinline__ int imin(int a, int b) { return a < b ? a : b; }
__device__ __forceinline__ int imax(int a, int b) { return a > b ? a : b; }

// ---------------------------------------------------------------------------
// generic fill (used for zeroing and for ctab = -1)
// ---------------------------------------------------------------------------
__global__ __launch_bounds__(256) void fill_u32_kernel(unsigned* p, unsigned v, int n) {
  int i = blockIdx.x * 256 + threadIdx.x;
  if (i < n) p[i] = v;
}

// ---------------------------------------------------------------------------
// per-batch max reduction (values are >= 0 so uint atomicMax on float bits)
// ---------------------------------------------------------------------------
__global__ __launch_bounds__(256) void amax_kernel(const float* __restrict__ x,
                                                   unsigned* out, int perB) {
  int b = blockIdx.y;
  const float* base = x + (size_t)b * perB;
  float m = 0.f;
  for (int i = blockIdx.x * blockDim.x + threadIdx.x; i < perB;
       i += gridDim.x * blockDim.x)
    m = fmaxf(m, base[i]);
  __shared__ float s[256];
  int tid = threadIdx.x;
  s[tid] = m; __syncthreads();
  for (int st = 128; st > 0; st >>= 1) {
    if (tid < st) s[tid] = fmaxf(s[tid], s[tid + st]);
    __syncthreads();
  }
  if (tid == 0) atomicMax(out + b, __float_as_uint(s[0]));
}

// ---------------------------------------------------------------------------
// normalize-by-max (or plain copy if scal==nullptr) into a channel slice of a
// wider buffer: out[b, coff+c, i] = in[b, c, i] / clamp(max_b, 1e-5, 1)
// ---------------------------------------------------------------------------
__global__ __launch_bounds__(256) void normcopy_kernel(const float* __restrict__ in,
                                                       float* __restrict__ out,
                                                       const unsigned* scal,
                                                       int B, int C, int HW,
                                                       int Ctot, int coff) {
  int idx = blockIdx.x * 256 + threadIdx.x;
  int total = B * C * HW;
  if (idx >= total) return;
  int b = idx / (C * HW);
  int rem = idx - b * C * HW;
  int c = rem / HW;
  int i = rem - c * HW;
  float v = in[idx];
  if (scal) {
    float s = __uint_as_float(scal[b]);
    s = fminf(fmaxf(s, 1e-5f), 1.0f);
    v /= s;
  }
  out[((size_t)(b * Ctot + coff + c)) * HW + i] = v;
}

// ---------------------------------------------------------------------------
// Implicit-GEMM conv via WMMA, fully compile-time configured.
//   M = 16 output channels, N = 16 output pixels per wave,
//   K = IC*KHW*KHW accumulated in unrolled chunks of 32 (f16 in, f32 acc).
// 4 waves / block.  LDS layouts:
//   sA[m][k]  (pitch padded)  -> A fragment = two contiguous 8-half runs
//   sB[n][k]  (pitch padded)  -> B fragment = two contiguous 8-half runs
// Staging is branchless (clamped addresses + ALU select) so global loads
// pipeline, and pair-packed so LDS stores are 32-bit.
// EDGE -> replicate-pad (reppad), else zero-pad.
// ---------------------------------------------------------------------------
template <int IC, int KHW, int STRIDE, int PAD, bool EDGE, int OC>
__global__ __launch_bounds__(128) void conv_wmma_kernel(
    const float* __restrict__ in, const float* __restrict__ w,
    const float* __restrict__ bias, float* __restrict__ out,
    int IH, int IW, int OH, int OW) {
  constexpr int Ktot = IC * KHW * KHW;
  constexpr int KC   = (Ktot + 31) / 32;
  constexpr int KPAD = KC * 32;
  constexpr int PITCH = KPAD + 8;            // 16B-aligned, bank-conflict pad

  __shared__ alignas(16) _Float16 sA[16 * PITCH];
  __shared__ alignas(16) _Float16 sB[4][16 * PITCH];

  const int tid  = threadIdx.x;
  const int wave = tid >> 5;
  const int lane = tid & 31;
  const int b    = blockIdx.y;

  // --- stage A (weights, OIHW flat == m*Ktot + k); branchless, pair-packed ---
  for (int i = tid; i < 8 * KPAD; i += 128) {
    int idx = i * 2;
    int m = idx / KPAD, k = idx - m * KPAD;   // compile-time divisor, k even
    int msafe = imin(m, OC - 1);
    float w0 = w[msafe * Ktot + imin(k, Ktot - 1)];
    float w1 = w[msafe * Ktot + imin(k + 1, Ktot - 1)];
    bool ok0 = (k < Ktot) && (m < OC);
    bool ok1 = (k + 1 < Ktot) && (m < OC);
    v2h pk;
    pk[0] = (_Float16)(ok0 ? w0 : 0.f);
    pk[1] = (_Float16)(ok1 ? w1 : 0.f);
    *(v2h*)&sA[m * PITCH + k] = pk;
  }
  __syncthreads();

  const int npx = OH * OW;
  const int tile_base = (blockIdx.x * 4 + wave) * 16;

  // --- stage B (im2col patch, N-major sB[n][k]); branchless, pair-packed ---
  for (int i = lane; i < 8 * KPAD; i += 32) {
    int idx = i * 2;
    int n = idx / KPAD, k = idx - n * KPAD;   // compile-time divisor, k even
    int gp = imin(tile_base + n, npx - 1);    // clamped; extra px harmless
    int oy = gp / OW, ox = gp - oy * OW;
    v2h pk;
#pragma unroll
    for (int e = 0; e < 2; ++e) {
      int kk = k + e;
      int c   = kk / (KHW * KHW);
      int rem = kk - c * (KHW * KHW);
      int ky  = rem / KHW, kx = rem - ky * KHW;
      int iy  = oy * STRIDE - PAD + ky;
      int ix  = ox * STRIDE - PAD + kx;
      bool ok = (kk < Ktot);
      if (!EDGE) ok = ok && (iy >= 0) && (iy < IH) && (ix >= 0) && (ix < IW);
      int iyc = imin(imax(iy, 0), IH - 1);
      int ixc = imin(imax(ix, 0), IW - 1);
      int cc  = imin(c, IC - 1);
      float v = in[(((size_t)b * IC + cc) * IH + iyc) * IW + ixc];
      pk[e] = (_Float16)(ok ? v : 0.f);
    }
    *(v2h*)&sB[wave][n * PITCH + k] = pk;
  }
  __syncthreads();

  v8f acc = {};
  const int grp  = lane >> 4;   // lane group (0: lanes 0-15, 1: lanes 16-31)
  const int mrow = lane & 15;

#pragma unroll
  for (int kc = 0; kc < KC; ++kc) {
    const int kbase = kc * 32;
    // A fragment: a[t] = A[m=mrow][kbase + (t<8 ? t : t+8) + 8*grp]
    const v8h a0 = *(const v8h*)&sA[mrow * PITCH + kbase + 8 * grp];
    const v8h a1 = *(const v8h*)&sA[mrow * PITCH + kbase + 16 + 8 * grp];
    // B fragment: bb[t] = B[k=kbase + (t<8 ? t : t+8) + 8*grp][n=mrow]
    const v8h b0 = *(const v8h*)&sB[wave][mrow * PITCH + kbase + 8 * grp];
    const v8h b1 = *(const v8h*)&sB[wave][mrow * PITCH + kbase + 16 + 8 * grp];
    v16h a, bb;
#pragma unroll
    for (int t = 0; t < 8; ++t) {
      a[t] = a0[t]; a[t + 8] = a1[t];
      bb[t] = b0[t]; bb[t + 8] = b1[t];
    }
    acc = __builtin_amdgcn_wmma_f32_16x16x32_f16(
        false, a, false, bb, (short)0, acc, false, false);
  }

  // --- store D: lane l reg r -> M = r + 8*grp, N = mrow ---
  int gp = tile_base + mrow;
  if (gp < npx) {
    int oy = gp / OW, ox = gp - oy * OW;
#pragma unroll
    for (int r = 0; r < 8; ++r) {
      int m = r + 8 * grp;
      if (m < OC)
        out[(((size_t)b * OC + m) * OH + oy) * OW + ox] = acc[r] + bias[m];
    }
  }
}

// ---------------------------------------------------------------------------
// GroupNorm (groups=2): stats then apply+relu
// ---------------------------------------------------------------------------
__global__ __launch_bounds__(256) void gn_stats_kernel(const float* __restrict__ x,
                                                       float* stats, int C, int HW,
                                                       int groups) {
  int bg  = blockIdx.y;            // b*groups + g; group slabs are contiguous
  int per = (C / groups) * HW;
  const float* base = x + (size_t)bg * per;
  float s = 0.f, ss = 0.f;
  for (int i = blockIdx.x * blockDim.x + threadIdx.x; i < per;
       i += gridDim.x * blockDim.x) {
    float v = base[i];
    s += v; ss += v * v;
  }
  __shared__ float sd[256];
  int tid = threadIdx.x;
  sd[tid] = s; __syncthreads();
  for (int st = 128; st > 0; st >>= 1) { if (tid < st) sd[tid] += sd[tid + st]; __syncthreads(); }
  if (tid == 0) atomicAdd(&stats[bg * 2], sd[0]);
  __syncthreads();
  sd[tid] = ss; __syncthreads();
  for (int st = 128; st > 0; st >>= 1) { if (tid < st) sd[tid] += sd[tid + st]; __syncthreads(); }
  if (tid == 0) atomicAdd(&stats[bg * 2 + 1], sd[0]);
}

__global__ __launch_bounds__(256) void gn_apply_relu_kernel(float* x, const float* stats,
                                                            const float* g, const float* be,
                                                            int B, int C, int HW, int groups) {
  int idx = blockIdx.x * 256 + threadIdx.x;
  int total = B * C * HW;
  if (idx >= total) return;
  int b = idx / (C * HW);
  int c = (idx / HW) % C;
  int grp = c / (C / groups);
  float per = (float)((C / groups) * HW);
  float mu  = stats[(b * groups + grp) * 2] / per;
  float var = stats[(b * groups + grp) * 2 + 1] / per - mu * mu;
  float v = (x[idx] - mu) * rsqrtf(var + 1e-5f) * g[c] + be[c];
  x[idx] = fmaxf(v, 0.f);
}

// ---------------------------------------------------------------------------
// bilinear resize (half-pixel centers) into channel slice of wider buffer
// ---------------------------------------------------------------------------
__global__ __launch_bounds__(256) void resize_kernel(const float* __restrict__ in,
                                                     float* __restrict__ out,
                                                     int B, int C, int IH, int IW,
                                                     int OH, int OW, int Ctot, int coff) {
  int idx = blockIdx.x * 256 + threadIdx.x;
  int total = B * C * OH * OW;
  if (idx >= total) return;
  int b  = idx / (C * OH * OW);
  int r0 = idx - b * C * OH * OW;
  int c  = r0 / (OH * OW);
  int r1 = r0 - c * (OH * OW);
  int oy = r1 / OW, ox = r1 - oy * OW;
  float sy = (float)IH / OH, sx = (float)IW / OW;
  float fy = (oy + 0.5f) * sy - 0.5f;
  float fx = (ox + 0.5f) * sx - 0.5f;
  int y0 = (int)floorf(fy), x0 = (int)floorf(fx);
  float wy = fy - y0, wx = fx - x0;
  int y0c = imin(imax(y0, 0), IH - 1), y1c = imin(imax(y0 + 1, 0), IH - 1);
  int x0c = imin(imax(x0, 0), IW - 1), x1c = imin(imax(x0 + 1, 0), IW - 1);
  const float* p = in + ((size_t)(b * C + c)) * IH * IW;
  float v = (1.f - wy) * ((1.f - wx) * p[y0c * IW + x0c] + wx * p[y0c * IW + x1c]) +
            wy * ((1.f - wx) * p[y1c * IW + x0c] + wx * p[y1c * IW + x1c]);
  out[(((size_t)(b * Ctot + coff + c)) * OH + oy) * OW + ox] = v;
}

// ---------------------------------------------------------------------------
// confidence: 0.5*(tanh+1) in place + global max; then normalize by max
// ---------------------------------------------------------------------------
__global__ __launch_bounds__(256) void tanh_max_kernel(float* x, unsigned* mx, int n) {
  int i = blockIdx.x * 256 + threadIdx.x;
  float v = 0.f;
  if (i < n) { v = 0.5f * (tanhf(x[i]) + 1.f); x[i] = v; }
  __shared__ float s[256];
  int tid = threadIdx.x;
  s[tid] = v; __syncthreads();
  for (int st = 128; st > 0; st >>= 1) { if (tid < st) s[tid] = fmaxf(s[tid], s[tid + st]); __syncthreads(); }
  if (tid == 0) atomicMax(mx, __float_as_uint(s[0]));
}

__global__ __launch_bounds__(256) void conf_write_kernel(const float* __restrict__ x,
                                                         float* __restrict__ out,
                                                         const unsigned* mx, int n) {
  int i = blockIdx.x * 256 + threadIdx.x;
  if (i >= n) return;
  float s = fmaxf(__uint_as_float(mx[0]), 1e-5f);
  out[i] = x[i] / s;
}

// ---------------------------------------------------------------------------
// bilateral solver: hash build / neighbors / splat / blur / CG
// ---------------------------------------------------------------------------
__device__ __forceinline__ int hash5(const int* c) {
  unsigned h = (unsigned)c[0] * 73856093u + (unsigned)c[1] * 19349663u +
               (unsigned)c[2] * 83492791u + (unsigned)c[3] * 49979687u +
               (unsigned)c[4] * 24036583u;
  return (int)(h & TMASK);
}

__global__ __launch_bounds__(256) void build_hash_kernel(const float* __restrict__ featb,
                                                         int* __restrict__ h_idx,
                                                         int* __restrict__ occ,
                                                         int* __restrict__ ctab,
                                                         int H, int W) {
  int i = blockIdx.x * 256 + threadIdx.x;
  int N = H * W;
  if (i >= N) return;
  int y = i / W, x = i - y * W;
  float r  = featb[i] * 255.f;
  float g  = featb[N + i] * 255.f;
  float bl = featb[2 * N + i] * 255.f;
  float Y = 0.299f * r + 0.587f * g + 0.114f * bl;
  float U = -0.168736f * r - 0.331264f * g + 0.5f * bl + 128.f;
  float V = 0.5f * r - 0.418688f * g - 0.081312f * bl + 128.f;
  int c[5];
  c[0] = (int)floorf((float)x / 7.0f);
  c[1] = (int)floorf((float)y / 7.0f);
  c[2] = (int)floorf(Y / 8.0f);
  c[3] = (int)floorf(U / 2.0f);
  c[4] = (int)floorf(V / 2.0f);
  int hv = hash5(c);
  h_idx[i] = hv;
  occ[hv] = 1;
#pragma unroll
  for (int d = 0; d < 5; ++d) ctab[hv * 5 + d] = c[d];
}

__global__ __launch_bounds__(256) void build_neighbors_kernel(const int* __restrict__ occ,
                                                              const int* __restrict__ ctab,
                                                              int* __restrict__ nhs,
                                                              unsigned* __restrict__ vb) {
  int t = blockIdx.x * 256 + threadIdx.x;
  if (t >= TSIZE) return;
  int c[5];
#pragma unroll
  for (int d = 0; d < 5; ++d) c[d] = ctab[t * 5 + d];
  unsigned bits = 0;
  int oc = occ[t];
#pragma unroll
  for (int d = 0; d < 5; ++d) {
#pragma unroll
    for (int s = 0; s < 2; ++s) {
      int nc[5];
#pragma unroll
      for (int d2 = 0; d2 < 5; ++d2) nc[d2] = c[d2];
      nc[d] += s ? 1 : -1;
      int nh = hash5(nc);
      int j = d * 2 + s;
      nhs[j * TSIZE + t] = nh;
      if (oc && occ[nh]) {
        bool eq = true;
#pragma unroll
        for (int d2 = 0; d2 < 5; ++d2) eq = eq && (ctab[nh * 5 + d2] == nc[d2]);
        if (eq) bits |= 1u << j;
      }
    }
  }
  vb[t] = bits;
}

__global__ __launch_bounds__(256) void splat_kernel(const float* __restrict__ predb,
                                                    const float* __restrict__ confb,
                                                    const int* __restrict__ h_idx,
                                                    float* m, float* wspl, float* bvec,
                                                    int N) {
  int i = blockIdx.x * 256 + threadIdx.x;
  if (i >= N) return;
  int hv = h_idx[i];
  float w = confb[i];
  atomicAdd(&m[hv], 1.f);
  atomicAdd(&wspl[hv], w);
  atomicAdd(&bvec[hv * 3 + 0], w * predb[i]);
  atomicAdd(&bvec[hv * 3 + 1], w * predb[N + i]);
  atomicAdd(&bvec[hv * 3 + 2], w * predb[2 * N + i]);
}

template <int C>
__global__ __launch_bounds__(256) void blur_kernel(const float* __restrict__ v,
                                                   float* __restrict__ out,
                                                   const int* __restrict__ occ,
                                                   const int* __restrict__ nhs,
                                                   const unsigned* __restrict__ vb) {
  int t = blockIdx.x * 256 + threadIdx.x;
  if (t >= TSIZE) return;
  if (!occ[t]) {
#pragma unroll
    for (int c = 0; c < C; ++c) out[t * C + c] = 0.f;
    return;
  }
  float acc[C];
#pragma unroll
  for (int c = 0; c < C; ++c) acc[c] = 10.f * v[t * C + c];  // 2*DIM = 10
  unsigned bits = vb[t];
#pragma unroll
  for (int j = 0; j < 10; ++j) {
    if (bits & (1u << j)) {
      int nh = nhs[j * TSIZE + t];
#pragma unroll
      for (int c = 0; c < C; ++c) acc[c] += v[nh * C + c];
    }
  }
#pragma unroll
  for (int c = 0; c < C; ++c) out[t * C + c] = acc[c];
}

__global__ __launch_bounds__(256) void occ_to_float_kernel(const int* occ, float* n) {
  int t = blockIdx.x * 256 + threadIdx.x;
  if (t < TSIZE) n[t] = occ[t] ? 1.f : 0.f;
}

__global__ __launch_bounds__(256) void bisto_update_kernel(float* n, const float* m,
                                                           const float* blurn) {
  int t = blockIdx.x * 256 + threadIdx.x;
  if (t < TSIZE) n[t] = sqrtf(n[t] * m[t] / (blurn[t] + EPS12));
}

__global__ __launch_bounds__(256) void mmul_kernel(float* m, const float* n,
                                                   const float* blurn) {
  int t = blockIdx.x * 256 + threadIdx.x;
  if (t < TSIZE) m[t] = n[t] * blurn[t];
}

__global__ __launch_bounds__(256) void init_sol_kernel(const float* bvec, const float* wspl,
                                                       const float* m, const float* n,
                                                       float* x, float* Minv) {
  int t = blockIdx.x * 256 + threadIdx.x;
  if (t >= TSIZE) return;
  float ws = wspl[t];
#pragma unroll
  for (int c = 0; c < 3; ++c) x[t * 3 + c] = bvec[t * 3 + c] / (ws + EPS12);
  float Ad = fmaxf(LAMBDA * (m[t] - 10.f * n[t] * n[t]) + ws, 1e-5f);
  Minv[t] = 1.f / Ad;
}

__global__ __launch_bounds__(256) void muln_kernel(const float* n, const float* v, float* o) {
  int t = blockIdx.x * 256 + threadIdx.x;
  if (t >= TSIZE) return;
  float nv = n[t];
#pragma unroll
  for (int c = 0; c < 3; ++c) o[t * 3 + c] = nv * v[t * 3 + c];
}

__global__ __launch_bounds__(256) void finishA_kernel(float* Ap, const float* v,
                                                      const float* m, const float* n,
                                                      const float* wspl) {
  int t = blockIdx.x * 256 + threadIdx.x;
  if (t >= TSIZE) return;
  float mv = m[t], nv = n[t], ws = wspl[t];
#pragma unroll
  for (int c = 0; c < 3; ++c)
    Ap[t * 3 + c] = LAMBDA * (mv * v[t * 3 + c] - nv * Ap[t * 3 + c]) + ws * v[t * 3 + c];
}

__global__ __launch_bounds__(256) void init_cg_kernel(const float* bvec, const float* Ax,
                                                      const float* Minv, float* r, float* z,
                                                      float* p) {
  int t = blockIdx.x * 256 + threadIdx.x;
  if (t >= TSIZE) return;
  float mi = Minv[t];
#pragma unroll
  for (int c = 0; c < 3; ++c) {
    float rv = bvec[t * 3 + c] - Ax[t * 3 + c];
    r[t * 3 + c] = rv;
    float zv = mi * rv;
    z[t * 3 + c] = zv;
    p[t * 3 + c] = zv;
  }
}

__global__ __launch_bounds__(256) void dot6_kernel(const float* r, const float* z,
                                                   const float* p, const float* Ap,
                                                   float* dots) {
  float s[6] = {0, 0, 0, 0, 0, 0};
  for (int t = blockIdx.x * blockDim.x + threadIdx.x; t < TSIZE;
       t += gridDim.x * blockDim.x) {
#pragma unroll
    for (int c = 0; c < 3; ++c) {
      s[c]     += r[t * 3 + c] * z[t * 3 + c];
      s[3 + c] += p[t * 3 + c] * Ap[t * 3 + c];
    }
  }
  __shared__ float sd[256];
  int tid = threadIdx.x;
#pragma unroll
  for (int k = 0; k < 6; ++k) {
    sd[tid] = s[k]; __syncthreads();
    for (int st = 128; st > 0; st >>= 1) { if (tid < st) sd[tid] += sd[tid + st]; __syncthreads(); }
    if (tid == 0) atomicAdd(&dots[k], sd[0]);
    __syncthreads();
  }
}

__global__ __launch_bounds__(256) void dot3_kernel(const float* r, const float* z,
                                                   float* dots3) {
  float s[3] = {0, 0, 0};
  for (int t = blockIdx.x * blockDim.x + threadIdx.x; t < TSIZE;
       t += gridDim.x * blockDim.x) {
#pragma unroll
    for (int c = 0; c < 3; ++c) s[c] += r[t * 3 + c] * z[t * 3 + c];
  }
  __shared__ float sd[256];
  int tid = threadIdx.x;
#pragma unroll
  for (int k = 0; k < 3; ++k) {
    sd[tid] = s[k]; __syncthreads();
    for (int st = 128; st > 0; st >>= 1) { if (tid < st) sd[tid] += sd[tid + st]; __syncthreads(); }
    if (tid == 0) atomicAdd(&dots3[k], sd[0]);
    __syncthreads();
  }
}

__global__ __launch_bounds__(256) void update1_kernel(float* x, float* r, float* z,
                                                      const float* p, const float* Ap,
                                                      const float* Minv, const float* dots) {
  int t = blockIdx.x * 256 + threadIdx.x;
  if (t >= TSIZE) return;
  float mi = Minv[t];
#pragma unroll
  for (int c = 0; c < 3; ++c) {
    float alpha = dots[c] / (dots[3 + c] + EPS12);
    x[t * 3 + c] += alpha * p[t * 3 + c];
    float rv = r[t * 3 + c] - alpha * Ap[t * 3 + c];
    r[t * 3 + c] = rv;
    z[t * 3 + c] = mi * rv;
  }
}

__global__ __launch_bounds__(256) void update2_kernel(float* p, const float* z,
                                                      const float* dots) {
  int t = blockIdx.x * 256 + threadIdx.x;
  if (t >= TSIZE) return;
#pragma unroll
  for (int c = 0; c < 3; ++c) {
    float beta = dots[6 + c] / (dots[c] + EPS12);
    p[t * 3 + c] = z[t * 3 + c] + beta * p[t * 3 + c];
  }
}

__global__ __launch_bounds__(256) void slice_kernel(const float* __restrict__ xv,
                                                    const int* __restrict__ h_idx,
                                                    float* __restrict__ outb, int N) {
  int i = blockIdx.x * 256 + threadIdx.x;
  if (i >= N) return;
  int hv = h_idx[i];
  outb[i]         = xv[hv * 3 + 0];
  outb[N + i]     = xv[hv * 3 + 1];
  outb[2 * N + i] = xv[hv * 3 + 2];
}

// ---------------------------------------------------------------------------
// host orchestration (all launches on `stream`; graph-capture safe)
// ---------------------------------------------------------------------------
extern "C" void kernel_launch(void* const* d_in, const int* in_sizes, int n_in,
                              void* d_out, int out_size, void* d_ws, size_t ws_size,
                              hipStream_t stream) {
  const float* image   = (const float*)d_in[0];
  const float* feature = (const float*)d_in[1];
  const float* pred    = (const float*)d_in[2];
  const float* w1  = (const float*)d_in[3];  const float* b1  = (const float*)d_in[4];
  const float* g1  = (const float*)d_in[5];  const float* be1 = (const float*)d_in[6];
  const float* w2  = (const float*)d_in[7];  const float* b2  = (const float*)d_in[8];
  const float* g2  = (const float*)d_in[9];  const float* be2 = (const float*)d_in[10];
  const float* wd1 = (const float*)d_in[11]; const float* bd1 = (const float*)d_in[12];
  const float* gd1 = (const float*)d_in[13]; const float* bed1= (const float*)d_in[14];
  const float* wd2 = (const float*)d_in[15]; const float* bd2 = (const float*)d_in[16];
  const float* gd2 = (const float*)d_in[17]; const float* bed2= (const float*)d_in[18];
  const float* wf  = (const float*)d_in[19]; const float* bf  = (const float*)d_in[20];
  (void)in_sizes; (void)n_in; (void)out_size; (void)ws_size;

  const int B = 2, H = 384, W = 512, H1 = 192, W1 = 256, H2 = 96, W2 = 128;
  const int HW = H * W, HW1 = H1 * W1, HW2 = H2 * W2;

  size_t off = 0;
  auto alloc = [&](size_t nf) -> float* { float* p = (float*)d_ws + off; off += nf; return p; };
  float* scalars = alloc(8);                       // [0..1] img max, [2..3] feat max, [4] conf max
  float* stats   = alloc(8);
  float* inp     = alloc((size_t)B * 6 * HW);
  float* featn   = alloc((size_t)B * 3 * HW);
  float* x1      = alloc((size_t)B * 16 * HW1);
  float* x2      = alloc((size_t)B * 16 * HW2);
  float* dx1     = alloc((size_t)B * 16 * HW2);
  float* cat     = alloc((size_t)B * 32 * HW1);
  float* dx2     = alloc((size_t)B * 16 * HW1);
  float* dx2r    = alloc((size_t)B * 16 * HW);
  float* confraw = alloc((size_t)B * HW);
  int*   h_idx   = (int*)alloc(HW);                // per-batch (reused)
  int*   occ     = (int*)alloc(TSIZE);
  int*   ctab    = (int*)alloc((size_t)5 * TSIZE);
  int*   nhs     = (int*)alloc((size_t)10 * TSIZE);
  unsigned* vb   = (unsigned*)alloc(TSIZE);
  float* mT      = alloc(TSIZE);
  float* nT      = alloc(TSIZE);
  float* tmp1    = alloc(TSIZE);
  float* wspl    = alloc(TSIZE);
  float* Minv    = alloc(TSIZE);
  float* bvec    = alloc((size_t)3 * TSIZE);
  float* xv      = alloc((size_t)3 * TSIZE);
  float* rv      = alloc((size_t)3 * TSIZE);
  float* zv      = alloc((size_t)3 * TSIZE);
  float* pv      = alloc((size_t)3 * TSIZE);
  float* Apv     = alloc((size_t)3 * TSIZE);
  float* t3      = alloc((size_t)3 * TSIZE);
  float* dots    = alloc(16);

  float* out_img  = (float*)d_out;
  float* out_conf = (float*)d_out + (size_t)B * 3 * HW;

  auto zero = [&](void* p, size_t n) {
    fill_u32_kernel<<<(int)((n + 255) / 256), 256, 0, stream>>>((unsigned*)p, 0u, (int)n);
  };
  const int TG = (TSIZE + 255) / 256;

  // ---- normalization + input assembly ----
  zero(scalars, 8);
  amax_kernel<<<dim3(64, B), 256, 0, stream>>>(image,   (unsigned*)scalars,     3 * HW);
  amax_kernel<<<dim3(64, B), 256, 0, stream>>>(feature, (unsigned*)scalars + 2, 3 * HW);
  {
    int tot = B * 3 * HW, gb = (tot + 255) / 256;
    normcopy_kernel<<<gb, 256, 0, stream>>>(image,   inp,   (unsigned*)scalars,     B, 3, HW, 6, 0);
    normcopy_kernel<<<gb, 256, 0, stream>>>(pred,    inp,   nullptr,                B, 3, HW, 6, 3);
    normcopy_kernel<<<gb, 256, 0, stream>>>(feature, featn, (unsigned*)scalars + 2, B, 3, HW, 3, 0);
  }

  // ---- conv1 (6->16, 4x4 s2, edge pad) + GN + relu ----
  conv_wmma_kernel<6, 4, 2, 1, true, 16>
      <<<dim3((HW1 + 63) / 64, B), 128, 0, stream>>>(inp, w1, b1, x1, H, W, H1, W1);
  zero(stats, 8);
  gn_stats_kernel<<<dim3(64, B * 2), 256, 0, stream>>>(x1, stats, 16, HW1, 2);
  gn_apply_relu_kernel<<<(B * 16 * HW1 + 255) / 256, 256, 0, stream>>>(x1, stats, g1, be1, B, 16, HW1, 2);

  // ---- conv2 (16->16, 4x4 s2, edge pad) + GN + relu ----
  conv_wmma_kernel<16, 4, 2, 1, true, 16>
      <<<dim3((HW2 + 63) / 64, B), 128, 0, stream>>>(x1, w2, b2, x2, H1, W1, H2, W2);
  zero(stats, 8);
  gn_stats_kernel<<<dim3(64, B * 2), 256, 0, stream>>>(x2, stats, 16, HW2, 2);
  gn_apply_relu_kernel<<<(B * 16 * HW2 + 255) / 256, 256, 0, stream>>>(x2, stats, g2, be2, B, 16, HW2, 2);

  // ---- wd1 (16->16, 3x3 s1, zero pad) + GN + relu ----
  conv_wmma_kernel<16, 3, 1, 1, false, 16>
      <<<dim3((HW2 + 63) / 64, B), 128, 0, stream>>>(x2, wd1, bd1, dx1, H2, W2, H2, W2);
  zero(stats, 8);
  gn_stats_kernel<<<dim3(64, B * 2), 256, 0, stream>>>(dx1, stats, 16, HW2, 2);
  gn_apply_relu_kernel<<<(B * 16 * HW2 + 255) / 256, 256, 0, stream>>>(dx1, stats, gd1, bed1, B, 16, HW2, 2);

  // ---- resize dx1 -> cat[:,0:16]; copy x1 -> cat[:,16:32] ----
  resize_kernel<<<(B * 16 * HW1 + 255) / 256, 256, 0, stream>>>(dx1, cat, B, 16, H2, W2, H1, W1, 32, 0);
  normcopy_kernel<<<(B * 16 * HW1 + 255) / 256, 256, 0, stream>>>(x1, cat, nullptr, B, 16, HW1, 32, 16);

  // ---- wd2 (32->16, 3x3 s1, zero pad) + GN + relu ----
  conv_wmma_kernel<32, 3, 1, 1, false, 16>
      <<<dim3((HW1 + 63) / 64, B), 128, 0, stream>>>(cat, wd2, bd2, dx2, H1, W1, H1, W1);
  zero(stats, 8);
  gn_stats_kernel<<<dim3(64, B * 2), 256, 0, stream>>>(dx2, stats, 16, HW1, 2);
  gn_apply_relu_kernel<<<(B * 16 * HW1 + 255) / 256, 256, 0, stream>>>(dx2, stats, gd2, bed2, B, 16, HW1, 2);

  // ---- resize dx2 -> full res; wf conv (16->1, 3x3, edge pad); confidence ----
  resize_kernel<<<(B * 16 * HW + 255) / 256, 256, 0, stream>>>(dx2, dx2r, B, 16, H1, W1, H, W, 16, 0);
  conv_wmma_kernel<16, 3, 1, 1, true, 1>
      <<<dim3((HW + 63) / 64, B), 128, 0, stream>>>(dx2r, wf, bf, confraw, H, W, H, W);
  tanh_max_kernel<<<(B * HW + 255) / 256, 256, 0, stream>>>(confraw, (unsigned*)scalars + 4, B * HW);
  conf_write_kernel<<<(B * HW + 255) / 256, 256, 0, stream>>>(confraw, out_conf, (unsigned*)scalars + 4, B * HW);

  // ---- bilateral solver per batch ----
  const int N = HW, NG = (N + 255) / 256;
  for (int b = 0; b < B; ++b) {
    const float* featb = featn + (size_t)b * 3 * HW;
    const float* predb = pred  + (size_t)b * 3 * HW;
    const float* confb = out_conf + (size_t)b * HW;

    zero(occ, TSIZE);
    fill_u32_kernel<<<(5 * TSIZE + 255) / 256, 256, 0, stream>>>((unsigned*)ctab, 0xFFFFFFFFu, 5 * TSIZE);
    build_hash_kernel<<<NG, 256, 0, stream>>>(featb, h_idx, occ, ctab, H, W);
    build_neighbors_kernel<<<TG, 256, 0, stream>>>(occ, ctab, nhs, vb);

    zero(mT, TSIZE); zero(wspl, TSIZE); zero(bvec, (size_t)3 * TSIZE);
    splat_kernel<<<NG, 256, 0, stream>>>(predb, confb, h_idx, mT, wspl, bvec, N);

    occ_to_float_kernel<<<TG, 256, 0, stream>>>(occ, nT);
    for (int it = 0; it < 10; ++it) {
      blur_kernel<1><<<TG, 256, 0, stream>>>(nT, tmp1, occ, nhs, vb);
      bisto_update_kernel<<<TG, 256, 0, stream>>>(nT, mT, tmp1);
    }
    blur_kernel<1><<<TG, 256, 0, stream>>>(nT, tmp1, occ, nhs, vb);
    mmul_kernel<<<TG, 256, 0, stream>>>(mT, nT, tmp1);     // m = n * blur(n)

    init_sol_kernel<<<TG, 256, 0, stream>>>(bvec, wspl, mT, nT, xv, Minv);
    // r0 = b - A(x0)
    muln_kernel<<<TG, 256, 0, stream>>>(nT, xv, t3);
    blur_kernel<3><<<TG, 256, 0, stream>>>(t3, Apv, occ, nhs, vb);
    finishA_kernel<<<TG, 256, 0, stream>>>(Apv, xv, mT, nT, wspl);
    init_cg_kernel<<<TG, 256, 0, stream>>>(bvec, Apv, Minv, rv, zv, pv);

    for (int it = 0; it < 12; ++it) {
      muln_kernel<<<TG, 256, 0, stream>>>(nT, pv, t3);
      blur_kernel<3><<<TG, 256, 0, stream>>>(t3, Apv, occ, nhs, vb);
      finishA_kernel<<<TG, 256, 0, stream>>>(Apv, pv, mT, nT, wspl);
      zero(dots, 16);
      dot6_kernel<<<1024, 256, 0, stream>>>(rv, zv, pv, Apv, dots);
      update1_kernel<<<TG, 256, 0, stream>>>(xv, rv, zv, pv, Apv, Minv, dots);
      dot3_kernel<<<1024, 256, 0, stream>>>(rv, zv, dots + 6);
      update2_kernel<<<TG, 256, 0, stream>>>(pv, zv, dots);
    }
    slice_kernel<<<NG, 256, 0, stream>>>(xv, h_idx, out_img + (size_t)b * 3 * HW, N);
  }
}
